// Encoder_54202487275779
// MI455X (gfx1250) — matrix operand order repaired and checked by reading, not compiled, not run on hardware
//
#include <hip/hip_runtime.h>
#include <hip/hip_bf16.h>

#define FEAT 128
#define KTOT 256   // combined K: [mean | x] @ [Wl ; Wr]^T

typedef __attribute__((ext_vector_type(16))) __bf16 v16bf;
typedef __attribute__((ext_vector_type(8)))  float  v8f;

// round-to-nearest-even f32 -> bf16 (bit manipulation)
__device__ __forceinline__ unsigned short f2bfu(float f) {
    unsigned int u = __builtin_bit_cast(unsigned int, f);
    u += 0x7FFFu + ((u >> 16) & 1u);
    return (unsigned short)(u >> 16);
}
__device__ __forceinline__ __bf16 bfbits(unsigned short s) {
    return __builtin_bit_cast(__bf16, s);
}

// ---------------- zero int workspace (deg + cursor contiguous) ---------
__global__ void zero_i32(int* __restrict__ p, int n) {
    int i = blockIdx.x * blockDim.x + threadIdx.x;
    int stride = gridDim.x * blockDim.x;
    for (; i < n; i += stride) p[i] = 0;
}

// ---------------- CSR build step 1: degree histogram -------------------
__global__ void deg_count(const long long* __restrict__ dst,
                          int* __restrict__ deg, int E) {
    int e = blockIdx.x * blockDim.x + threadIdx.x;
    if (e >= E) return;
    atomicAdd(&deg[(int)dst[e]], 1);
}

// ---------------- CSR build step 2: exclusive scan (one workgroup) -----
__global__ __launch_bounds__(1024) void scan_rowptr(const int* __restrict__ deg,
                                                    int* __restrict__ rowptr, int N) {
    __shared__ int part[1024];
    int tid   = threadIdx.x;
    int chunk = (N + 1023) / 1024;
    int beg = tid * chunk;
    int end = beg + chunk; if (end > N) end = N;
    int s = 0;
    for (int i = beg; i < end; ++i) s += deg[i];
    part[tid] = s;
    __syncthreads();
    // Hillis-Steele inclusive scan over 1024 partials
    for (int off = 1; off < 1024; off <<= 1) {
        int v = (tid >= off) ? part[tid - off] : 0;
        __syncthreads();
        part[tid] += v;
        __syncthreads();
    }
    int run = (tid == 0) ? 0 : part[tid - 1];   // exclusive prefix of chunk
    for (int i = beg; i < end; ++i) { rowptr[i] = run; run += deg[i]; }
    if (tid == 1023) rowptr[N] = part[1023];    // total edge count
}

// ---------------- CSR build step 3: bucket fill ------------------------
__global__ void fill_csr(const long long* __restrict__ src,
                         const long long* __restrict__ dst,
                         const int* __restrict__ rowptr,
                         int* __restrict__ cursor,
                         int* __restrict__ csr, int E) {
    int e = blockIdx.x * blockDim.x + threadIdx.x;
    if (e >= E) return;
    int d = (int)dst[e];
    int pos = atomicAdd(&cursor[d], 1);
    csr[rowptr[d] + pos] = (int)src[e];
}

// ---------------- per-layer gather mean: one wave per node -------------
// mean[i][:] = sum_{j in N(i)} x[j][:] / max(deg(i),1)   (no atomics)
__global__ __launch_bounds__(256) void gather_mean(const int* __restrict__ rowptr,
                                                   const int* __restrict__ csr,
                                                   const float* __restrict__ xin,
                                                   float* __restrict__ mean, int N) {
    int wid  = (blockIdx.x * 256 + threadIdx.x) >> 5;   // node id (wave32)
    int part = threadIdx.x & 31;                        // 4 features per lane
    if (wid >= N) return;
    int b = rowptr[wid], e = rowptr[wid + 1];
    float4 acc = {0.f, 0.f, 0.f, 0.f};
    for (int j = b; j < e; ++j) {
        int s = csr[j];
        if (j + 1 < e)   // hide L2 latency on the next neighbor row
            __builtin_prefetch(xin + (long long)csr[j + 1] * FEAT + part * 4, 0, 0);
        float4 v = ((const float4*)(xin + (long long)s * FEAT))[part];
        acc.x += v.x; acc.y += v.y; acc.z += v.z; acc.w += v.w;
    }
    int deg = e - b;
    float inv = 1.0f / (float)(deg > 1 ? deg : 1);
    float4 m = {acc.x * inv, acc.y * inv, acc.z * inv, acc.w * inv};
    ((float4*)(mean + (long long)wid * FEAT))[part] = m;
}

// ---------------- fused dual-linear (WMMA) + bias + PReLU --------------
// out[r][:] = prelu( mean[r] @ Wl^T + bl + x[r] @ Wr^T, alpha )
// Block = 256 threads = 8 wave32 waves; each wave: 16 rows x 128 cols.
__global__ __launch_bounds__(256) void sage_gemm_prelu(
    const float* __restrict__ mean, const float* __restrict__ xin,
    const float* __restrict__ Wl,  const float* __restrict__ bl,
    const float* __restrict__ Wr,  const float* __restrict__ alpha,
    float* __restrict__ out, int N) {

    // combined weight W[o][k] (k<128: Wl, k>=128: Wr), bf16 in LDS, [o][KTOT]
    __shared__ unsigned short sW[FEAT * KTOT];   // 64 KB of 320 KB WGP LDS

    int tid = threadIdx.x;
    for (int i = tid; i < FEAT * FEAT; i += 256) {
        int o = i >> 7;
        int k = i & 127;
        sW[o * KTOT + k]        = f2bfu(Wl[i]);
        sW[o * KTOT + FEAT + k] = f2bfu(Wr[i]);
    }
    __syncthreads();

    int wave = tid >> 5;
    int lane = tid & 31;
    int half = lane >> 4;
    int l16  = lane & 15;
    int rowBase = (blockIdx.x * 8 + wave) * 16;

    int r   = rowBase + l16;            // A row (both lane-halves hold M=lane&15)
    bool rv = (r < N);
    const float* mRow = mean + (long long)(rv ? r : 0) * FEAT;
    const float* xRow = xin  + (long long)(rv ? r : 0) * FEAT;

    v8f acc[8] = {};

    #pragma unroll
    for (int ks = 0; ks < 8; ++ks) {
        int kk = ks * 32;
        // A fragment: K = {k0..k0+15}, k0 = kk + 16*half
        v16bf afrag;
        int k0 = kk + half * 16;
        #pragma unroll
        for (int seg = 0; seg < 2; ++seg) {
            int kb = k0 + seg * 8;
            #pragma unroll
            for (int j = 0; j < 8; ++j) {
                int k = kb + j;
                float f = (k < FEAT) ? mRow[k] : xRow[k - FEAT];
                f = rv ? f : 0.0f;      // value-select; EXEC stays full for WMMA
                afrag[seg * 8 + j] = bfbits(f2bfu(f));
            }
        }
        // 8 column tiles: B from LDS, WMMA accumulate
        #pragma unroll
        for (int t = 0; t < 8; ++t) {
            int o  = t * 16 + l16;
            int kb = kk + half * 16;
            const unsigned short* wp = &sW[o * KTOT + kb];
            v16bf bfrag;
            #pragma unroll
            for (int j = 0; j < 16; ++j) bfrag[j] = bfbits(wp[j]);
            acc[t] = __builtin_amdgcn_wmma_f32_16x16x32_bf16(
                false, afrag, false, bfrag, (short)0, acc[t], false, false);
        }
    }

    // epilogue: D lane layout (lane<16: M=v, N=lane; lane>=16: M=8+v, N=lane-16)
    #pragma unroll
    for (int t = 0; t < 8; ++t) {
        int n = t * 16 + l16;
        float b  = bl[n];
        float al = alpha[n];
        #pragma unroll
        for (int v = 0; v < 8; ++v) {
            int rr = rowBase + half * 8 + v;
            if (rr < N) {
                float val = acc[t][v] + b;
                val = (val >= 0.0f) ? val : al * val;
                out[(long long)rr * FEAT + n] = val;
            }
        }
    }
}

extern "C" void kernel_launch(void* const* d_in, const int* in_sizes, int n_in,
                              void* d_out, int out_size, void* d_ws, size_t ws_size,
                              hipStream_t stream) {
    const float*     x  = (const float*)d_in[0];
    const long long* ei = (const long long*)d_in[1];   // int64 [2, E]
    const float* Wl[3] = {(const float*)d_in[2],  (const float*)d_in[6],  (const float*)d_in[10]};
    const float* bL[3] = {(const float*)d_in[3],  (const float*)d_in[7],  (const float*)d_in[11]};
    const float* Wr[3] = {(const float*)d_in[4],  (const float*)d_in[8],  (const float*)d_in[12]};
    const float* aP[3] = {(const float*)d_in[5],  (const float*)d_in[9],  (const float*)d_in[13]};

    const int N = in_sizes[0] / FEAT;     // 100000
    const int E = in_sizes[1] / 2;        // 1600000

    // workspace layout (floats then ints):
    // [mean N*128][h1 N*128][h2 N*128][deg N][cursor N][rowptr N+1][csr E]
    float* mean = (float*)d_ws;
    float* h1   = mean + (size_t)N * FEAT;
    float* h2   = h1   + (size_t)N * FEAT;
    int*   deg    = (int*)(h2 + (size_t)N * FEAT);
    int*   cursor = deg + N;
    int*   rowptr = cursor + N;
    int*   csr    = rowptr + (N + 1);

    const int eBlocks = (E + 255) / 256;
    const int gBlocks = (N + 127) / 128;               // 8 waves x 16 rows / block
    const int mBlocks = (N * 32 + 255) / 256;          // one wave per node

    // ---- build CSR once (graph shared by all 3 layers) ----
    zero_i32<<<1024, 256, 0, stream>>>(deg, 2 * N);    // deg + cursor contiguous
    deg_count<<<eBlocks, 256, 0, stream>>>(ei + E, deg, E);
    scan_rowptr<<<1, 1024, 0, stream>>>(deg, rowptr, N);
    fill_csr<<<eBlocks, 256, 0, stream>>>(ei, ei + E, rowptr, cursor, csr, E);

    // ---- 3 layers ----
    const float* cur = x;
    float* outs[3] = {h1, h2, (float*)d_out};
    for (int L = 0; L < 3; ++L) {
        gather_mean<<<mBlocks, 256, 0, stream>>>(rowptr, csr, cur, mean, N);
        sage_gemm_prelu<<<gBlocks, 256, 0, stream>>>(mean, cur,
                                                     Wl[L], bL[L], Wr[L], aP[L],
                                                     outs[L], N);
        cur = outs[L];
    }
}
